// QuantizedLinear_37864431681975
// MI455X (gfx1250) — matrix operand order repaired and checked by reading, not compiled, not run on hardware
//
#include <hip/hip_runtime.h>

// ---------------------------------------------------------------------------
// W8A16 grouped-quant GEMM for MI455X (gfx1250, wave32, WMMA + TDM).
// out[32,16384] = x[32,4096]f16 @ (int8 W * per-(o,group) f16 scale).T + bias
// Weight-bandwidth bound (64 MiB int8 / 23.3 TB/s ~= 2.9us). Strategy:
//   - Tensor Data Mover streams W/x tiles global->LDS (padded rows),
//     quad-buffered with 2-chunk prefetch depth, one barrier per chunk
//   - register dequant via perm/pk-f16 bit trick (1.25 VALU per byte-lane)
//   - v_wmma_f32_16x16x32_f16, per-group scale folded into f32 accumulators
// ---------------------------------------------------------------------------

typedef __attribute__((ext_vector_type(16))) _Float16 v16h;
typedef __attribute__((ext_vector_type(8)))  float    v8f;
typedef __attribute__((ext_vector_type(2)))  _Float16 h2;
typedef __attribute__((ext_vector_type(4)))  int      i32x4;
typedef __attribute__((ext_vector_type(8)))  int      i32x8;
typedef __attribute__((ext_vector_type(4)))  unsigned u32x4;

#define K_DIM    4096
#define N_DIM    16384
#define B_M      32
#define GS       128                 // quant group size == K chunk
#define NG       (K_DIM / GS)        // 32 groups
#define NCHUNK   NG
#define NTILE    64                  // output channels per block
#define NTHREADS 128                 // 4 waves of 32
#define WSTRIDE  144                 // LDS weight row stride (128 + 16 pad) bytes
#define XSTRIDE  272                 // LDS x row stride (256 + 16 pad) bytes
#define WBUF_SZ  (NTILE * WSTRIDE)   // 9216 B
#define XBUF_SZ  (B_M * XSTRIDE)     // 8704 B

#define TDM_AVAILABLE __has_builtin(__builtin_amdgcn_tensor_load_to_lds)

#if TDM_AVAILABLE
#define NBUF 4                       // quad buffer: depth-2 TDM prefetch
#else
#define NBUF 2
#endif
#define SCL_OFF  (NBUF * (WBUF_SZ + XBUF_SZ))
#define SCL_SZ   (NTILE * NG * 2)    // 4096 B
#define LDS_SZ   (SCL_OFF + SCL_SZ)
#define XBUF_OFF (NBUF * WBUF_SZ)

// fp16x2 subtract 1152.0 on a bit-packed dword (v_pk_add_f16 w/ inline const)
static __device__ __forceinline__ unsigned pk_sub_1152(unsigned a)
{
    h2 t;
    __builtin_memcpy(&t, &a, 4);
    t = t - (h2){(_Float16)1152.0f, (_Float16)1152.0f};
    unsigned r;
    __builtin_memcpy(&r, &t, 4);
    return r;
}

#if TDM_AVAILABLE
// Issue a 2D TDM tile load global->LDS with per-row LDS padding.
// Dims/strides in bytes (data_size = 1B). pad codes per D# group1 encoding:
// pad_interval: 0=2,1=4,...,7=256 DWORDs ; pad_amount: n -> n+1 DWORDs
static __device__ __forceinline__ void tdm_load_2d(
    unsigned lds_addr, const void* gaddr,
    unsigned tensor_d0, unsigned tensor_d1,
    unsigned tile_d0,  unsigned tile_d1,
    unsigned stride0,
    unsigned pad_interval_code, unsigned pad_amount_code)
{
    const unsigned long long ga = (unsigned long long)(uintptr_t)gaddr;

    u32x4 g0;
    g0[0] = 1u;                                                  // count=1, user D#
    g0[1] = lds_addr;                                            // lds_addr
    g0[2] = (unsigned)(ga & 0xFFFFFFFFu);                        // global_addr[31:0]
    g0[3] = ((unsigned)(ga >> 32) & 0x01FFFFFFu) | 0x80000000u;  // addr[56:32]|type=2

    // group1 bitfield packing (see CDNA5 ISA)
    const unsigned w0 = (1u << 20)                      // pad_enable
                      | (pad_interval_code << 22)
                      | (pad_amount_code << 25);        // data_size=1B, mask=0
    const unsigned w1 = (tensor_d0 & 0xFFFFu) << 16;    // tensor_dim0[15:0] @ bits63:48
    const unsigned w2 = (tensor_d0 >> 16) | ((tensor_d1 & 0xFFFFu) << 16);
    const unsigned w3 = (tensor_d1 >> 16) | (tile_d0 << 16);     // tile_dim0 @127:112
    const unsigned w4 = tile_d1;                        // tile_dim1 ; tile_dim2=0
    const unsigned w5 = stride0;                        // tensor_dim0_stride[31:0]
    const i32x8 g1 = { (int)w0, (int)w1, (int)w2, (int)w3,
                       (int)w4, (int)w5, 0, 0 };
    const i32x4 gz = { 0, 0, 0, 0 };
#if __clang_major__ >= 23
    const i32x8 gz8 = { 0, 0, 0, 0, 0, 0, 0, 0 };
    __builtin_amdgcn_tensor_load_to_lds(g0, g1, gz, gz, gz8, 0);
#else
    __builtin_amdgcn_tensor_load_to_lds(g0, g1, gz, gz, 0);
#endif
}
#endif // TDM_AVAILABLE

__global__ __launch_bounds__(NTHREADS)
void qlin_wmma_kernel(const _Float16* __restrict__ x,
                      const signed char* __restrict__ qw,
                      const _Float16* __restrict__ scales,
                      const _Float16* __restrict__ bias,
                      _Float16* __restrict__ out)
{
    __shared__ unsigned char lds[LDS_SZ];

    const int tid  = threadIdx.x;
    const int wave = tid >> 5;
    const int lane = tid & 31;
    const int n0   = blockIdx.x * NTILE;

    const int col  = lane & 15;          // owned column within wave's N tile
    const int half = lane >> 4;          // lane-half (K-offset selector)

    v8f acc0 = {}, acc1 = {};            // running f32 accumulators (2 M-tiles)

    // ---- stage this block's scale tile into LDS (64ch x 32grp fp16, 4KiB) ----
    {
        const unsigned char* sg = (const unsigned char*)(scales + (size_t)n0 * NG);
        *(i32x4*)&lds[SCL_OFF + tid * 32]      = *(const i32x4*)(sg + tid * 32);
        *(i32x4*)&lds[SCL_OFF + tid * 32 + 16] = *(const i32x4*)(sg + tid * 32 + 16);
    }

#if TDM_AVAILABLE
    // LDS base address derived from the real array => array escapes, so the
    // compiler must assume the TDM engine writes it.
    const unsigned lds_base = (unsigned)(uintptr_t)(void*)&lds[0];

    const signed char*   wgb = qw + (size_t)n0 * K_DIM;
    const unsigned char* xgb = (const unsigned char*)x;

    if (wave == 0) {
        // prime the pipeline: chunks 0 and 1
#pragma unroll
        for (int p = 0; p < 2; ++p) {
            tdm_load_2d(lds_base + (unsigned)p * WBUF_SZ,
                        wgb + (size_t)p * GS,
                        K_DIM, N_DIM, GS, NTILE, K_DIM, 4, 3);
            tdm_load_2d(lds_base + XBUF_OFF + (unsigned)p * XBUF_SZ,
                        xgb + (size_t)p * (2 * GS),
                        2 * K_DIM, B_M, 2 * GS, B_M, 2 * K_DIM, 5, 3);
        }
    }
#else
    // ---------------- fallback: register-staged global->LDS ----------------
    const int wrow = tid >> 1;
    const int wcol = (tid & 1) * 64;
    const int xrow = tid >> 2;
    const int xcol = (tid & 3) * 64;
    const signed char*   wg = qw + (size_t)(n0 + wrow) * K_DIM + wcol;
    const unsigned char* xg = (const unsigned char*)x + (size_t)xrow * (K_DIM * 2) + xcol;
    i32x4 wst[4], xst[4];
#pragma unroll
    for (int i = 0; i < 4; ++i) {
        wst[i] = *(const i32x4*)(wg + 16 * i);
        xst[i] = *(const i32x4*)(xg + 16 * i);
    }
#endif

#pragma clang loop unroll(disable)
    for (int c = 0; c < NCHUNK; ++c) {
        unsigned char* wb = &lds[(c & (NBUF - 1)) * WBUF_SZ];
        unsigned char* xb = &lds[XBUF_OFF + (c & (NBUF - 1)) * XBUF_SZ];

#if TDM_AVAILABLE
        if (wave == 0) {
            if (c + 2 < NCHUNK) {
                // issue chunk c+2; its buffer was last read at compute(c-2),
                // already sequenced by the barrier chain (quad buffer).
                const unsigned bsel = (unsigned)((c + 2) & (NBUF - 1));
                tdm_load_2d(lds_base + bsel * WBUF_SZ, wgb + (size_t)(c + 2) * GS,
                            K_DIM, N_DIM, GS, NTILE, K_DIM, 4, 3);
                tdm_load_2d(lds_base + XBUF_OFF + bsel * XBUF_SZ,
                            xgb + (size_t)(c + 2) * (2 * GS),
                            2 * K_DIM, B_M, 2 * GS, B_M, 2 * K_DIM, 5, 3);
                __builtin_amdgcn_s_wait_tensorcnt(4);   // pair c retired (in-order)
            } else if (c + 1 < NCHUNK) {
                __builtin_amdgcn_s_wait_tensorcnt(2);   // pair c retired
            } else {
                __builtin_amdgcn_s_wait_tensorcnt(0);
            }
        }
        __syncthreads();                 // chunk c visible to all waves
        // TDM wrote LDS behind the compiler's back: force reloads.
        asm volatile("" : : "r"(&lds[0]) : "memory");
#else
#pragma unroll
        for (int i = 0; i < 4; ++i) {
            *(i32x4*)(wb + wrow * WSTRIDE + wcol + 16 * i) = wst[i];
            *(i32x4*)(xb + xrow * XSTRIDE + xcol + 16 * i) = xst[i];
        }
        __syncthreads();
        if (c + 1 < NCHUNK) {
#pragma unroll
            for (int i = 0; i < 4; ++i) {
                wst[i] = *(const i32x4*)(wg + (size_t)(c + 1) * GS + 16 * i);
                xst[i] = *(const i32x4*)(xg + (size_t)(c + 1) * (GS * 2) + 16 * i);
            }
        }
#endif

        // ------------------------- compute chunk c -------------------------
        v8f c0 = {}, c1 = {};

        const unsigned char* wrowp  = wb + (wave * 16 + col) * WSTRIDE + half * 16;
        const unsigned char* xrowp0 = xb + col * XSTRIDE;        // M tile 0 rows 0..15
        const unsigned char* xrowp1 = xrowp0 + 16 * XSTRIDE;     // M tile 1 rows 16..31
        const int ahoff = half * 32;                             // (lane>=16 ? +16 elems)

#pragma unroll
        for (int ks = 0; ks < 4; ++ks) {
            const int kb = ks * 32;      // K position within group (byte == element)

            // B fragment: 16 contiguous int8 of this column -> 16 halves.
            // u = q^0x80; (0x6400|u) == fp16(1024+u); -1152 -> exact q.
            union { i32x4 v; unsigned u[4]; } qb;
            qb.v = *(const i32x4*)(wrowp + kb);                  // ds_load_b128
            union { v16h h; unsigned u[8]; } bf;
#pragma unroll
            for (int j = 0; j < 4; ++j) {
                const unsigned uu = qb.u[j] ^ 0x80808080u;
                const unsigned lo = __builtin_amdgcn_perm(0x64646464u, uu, 0x05010400u);
                const unsigned hi = __builtin_amdgcn_perm(0x64646464u, uu, 0x05030402u);
                bf.u[2 * j]     = pk_sub_1152(lo);
                bf.u[2 * j + 1] = pk_sub_1152(hi);
            }

            // A fragments, contiguous K layout (K {0..15} / +16)
            union { v16h h; i32x4 d[2]; } a0, a1;
            a0.d[0] = *(const i32x4*)(xrowp0 + kb * 2 + ahoff);
            a0.d[1] = *(const i32x4*)(xrowp0 + kb * 2 + ahoff + 16);
            a1.d[0] = *(const i32x4*)(xrowp1 + kb * 2 + ahoff);
            a1.d[1] = *(const i32x4*)(xrowp1 + kb * 2 + ahoff + 16);

            c0 = __builtin_amdgcn_wmma_f32_16x16x32_f16(false, a0.h, false, bf.h,
                                                        (short)0, c0, false, false);
            c1 = __builtin_amdgcn_wmma_f32_16x16x32_f16(false, a1.h, false, bf.h,
                                                        (short)0, c1, false, false);
        }

        // fold this group's scale (from LDS tile) into the running accumulators
        const _Float16* sl = (const _Float16*)&lds[SCL_OFF];
        const float s = (float)sl[(wave * 16 + col) * NG + c];
#pragma unroll
        for (int i = 0; i < 8; ++i) {
            acc0[i] += s * c0[i];
            acc1[i] += s * c1[i];
        }
    }

    // ---- epilogue: bias + fp16 store (C layout: lane half selects M base) ----
    const int   o  = n0 + wave * 16 + col;
    const float bv = (float)bias[o];
    const int   mb = half * 8;
#pragma unroll
    for (int i = 0; i < 8; ++i) {
        out[(size_t)(mb + i) * N_DIM + o]      = (_Float16)(acc0[i] + bv);
        out[(size_t)(16 + mb + i) * N_DIM + o] = (_Float16)(acc1[i] + bv);
    }
}

extern "C" void kernel_launch(void* const* d_in, const int* in_sizes, int n_in,
                              void* d_out, int out_size, void* d_ws, size_t ws_size,
                              hipStream_t stream)
{
    (void)in_sizes; (void)n_in; (void)out_size; (void)d_ws; (void)ws_size;
    const _Float16*    xp = (const _Float16*)d_in[0];
    const signed char* wp = (const signed char*)d_in[1];
    const _Float16*    sp = (const _Float16*)d_in[2];
    const _Float16*    bp = (const _Float16*)d_in[3];
    _Float16*          op = (_Float16*)d_out;

    qlin_wmma_kernel<<<dim3(N_DIM / NTILE), NTHREADS, 0, stream>>>(xp, wp, sp, bp, op);
}
